// SelfAttention_90898687852620
// MI455X (gfx1250) — compile-verified, matrix operand order checked by reading, not run on hardware
//
#include <hip/hip_runtime.h>
#include <math.h>
#include <stdint.h>

// ---------------------------------------------------------------------------
// Causal self-attention head for MI455X (gfx1250), bf16 WMMA pipeline.
//   x:[512,256,384] f32, Wq/Wk/Wv:[384,384] f32 -> out:[512,256,384] f32
// Kernel 1: q,k (bf16 row-major) and v (bf16 transposed per batch) into ws.
// Kernel 2: flash attention with v_wmma_f32_16x16x32_bf16, async-to-LDS
//           staging (ASYNCcnt) and L2 prefetch of the next tile.
// ---------------------------------------------------------------------------

typedef __bf16  v16bf __attribute__((ext_vector_type(16)));
typedef __bf16  v8bf  __attribute__((ext_vector_type(8)));
typedef float   v8f   __attribute__((ext_vector_type(8)));
typedef float   v4f   __attribute__((ext_vector_type(4)));

#define NB   512
#define TT   256
#define CC   384
#define HH   384
#define NTOK (NB * TT)

__device__ __forceinline__ float exp2_fast(float x) {
#if defined(__HIP_DEVICE_COMPILE__) && __has_builtin(__builtin_amdgcn_exp2f)
  return __builtin_amdgcn_exp2f(x);   // v_exp_f32
#else
  return exp2f(x);                    // host-pass semantic check only
#endif
}

__device__ __forceinline__ v16bf cat8(v8bf a, v8bf b) {
  return __builtin_shufflevector(a, b, 0,1,2,3,4,5,6,7,8,9,10,11,12,13,14,15);
}

__device__ __forceinline__ v8f wmma_bf16(v16bf a, v16bf b, v8f c) {
  // D = A(16x32) * B(32x16) + C(16x16), f32 accumulate
  return __builtin_amdgcn_wmma_f32_16x16x32_bf16(false, a, false, b,
                                                 (short)0, c, false, false);
}

// Async DMA: 16 bytes global -> LDS per active lane, tracked by ASYNCcnt.
__device__ __forceinline__ void async_ld16(void* lds_ptr, const void* gptr) {
#if defined(__HIP_DEVICE_COMPILE__)
  unsigned lds_off = (unsigned)(uintptr_t)lds_ptr;   // addr[31:0] == LDS offset
  asm volatile("global_load_async_to_lds_b128 %0, %1, off"
               :
               : "v"(lds_off), "v"(gptr)
               : "memory");
#else
  (void)lds_ptr; (void)gptr;
#endif
}

__device__ __forceinline__ void wait_async0() {
#if defined(__HIP_DEVICE_COMPILE__)
  asm volatile("s_wait_asynccnt 0x0" ::: "memory");
#endif
}

// ---------------------------------------------------------------------------
// Kernel 1: QKV projection. grid = (NTOK/64, 3), block = 256 (8 waves).
//   blockIdx.y: 0=Q, 1=K, 2=V.  Wave w: token tile (w&3), feature half (w>>2).
// ---------------------------------------------------------------------------
__global__ __launch_bounds__(256) void qkv_proj_kernel(
    const float* __restrict__ x,
    const float* __restrict__ Wq, const float* __restrict__ Wk,
    const float* __restrict__ Wv,
    __bf16* __restrict__ qg, __bf16* __restrict__ kg, __bf16* __restrict__ vT)
{
  __shared__ __align__(16) __bf16 WtLds[CC * 40];   // [out-feat][k-chunk 32], 30720 B

  const int m    = blockIdx.y;
  const float* W = (m == 0) ? Wq : (m == 1) ? Wk : Wv;
  const int tok0 = blockIdx.x * 64;
  const int tid  = threadIdx.x;
  const int wave = tid >> 5;
  const int lane = tid & 31;
  const int l15  = lane & 15;
  const int hi16 = lane >> 4;
  const int t16  = tok0 + (wave & 3) * 16;    // this wave's 16-token tile
  const int fh   = (wave >> 2) * 192;         // this wave's feature half

  v8f acc[12];
#pragma unroll
  for (int j = 0; j < 12; ++j) acc[j] = (v8f)0.0f;

  for (int kk = 0; kk < CC; kk += 32) {
    // Stage W[kk..kk+31][0..383] transposed as bf16 into LDS (coalesced reads,
    // f32->bf16 conversion in flight, so this stays on the VALU path).
    for (int idx = tid; idx < 32 * CC; idx += 256) {
      int k = idx / CC, n = idx % CC;
      WtLds[n * 40 + k] = (__bf16)W[(size_t)(kk + k) * HH + n];
    }
    __syncthreads();

    // A fragment: 16 tokens x 32 features of x, f32 -> bf16.
    v16bf a;
    {
      const float* xp = x + (size_t)(t16 + l15) * CC + kk + hi16 * 8;
#pragma unroll
      for (int g = 0; g < 2; ++g) {
        v4f f0 = *(const v4f*)(xp + g * 16);
        v4f f1 = *(const v4f*)(xp + g * 16 + 4);
#pragma unroll
        for (int e = 0; e < 4; ++e) {
          a[g * 8 + e]     = (__bf16)f0[e];
          a[g * 8 + 4 + e] = (__bf16)f1[e];
        }
      }
    }

    // Double-buffered B-fragment stream from LDS.
    const __bf16* wb = &WtLds[(fh + l15) * 40 + hi16 * 8];
    v16bf b = cat8(*(const v8bf*)wb, *(const v8bf*)(wb + 16));
#pragma unroll
    for (int j = 0; j < 12; ++j) {
      v16bf bn = b;
      if (j < 11) {
        const __bf16* bp = wb + (size_t)(j + 1) * 16 * 40;
        bn = cat8(*(const v8bf*)bp, *(const v8bf*)(bp + 16));
      }
      acc[j] = wmma_bf16(a, b, acc[j]);
      b = bn;
    }
    __syncthreads();
  }

  // Write results: Q/K row-major bf16; V transposed per batch [b][h][t].
#pragma unroll
  for (int j = 0; j < 12; ++j) {
#pragma unroll
    for (int r = 0; r < 8; ++r) {
      int tok = t16 + r + hi16 * 8;
      int f   = fh + j * 16 + l15;
      __bf16 vv = (__bf16)acc[j][r];
      if (m == 0) {
        qg[(size_t)tok * HH + f] = vv;
      } else if (m == 1) {
        kg[(size_t)tok * HH + f] = vv;
      } else {
        int b2 = tok >> 8, tl = tok & 255;
        vT[((size_t)b2 * HH + f) * TT + tl] = vv;
      }
    }
  }
}

// ---------------------------------------------------------------------------
// Kernel 2: causal flash attention. grid = (T/64, B), block = 256 (8 waves).
//   Wave pair p = wave>>1 owns query tile qbase + p*16; half h = wave&1 owns
//   192 of the 384 output features. Both waves of a pair compute S redundantly.
// ---------------------------------------------------------------------------
__global__ __launch_bounds__(256) void flash_attn_kernel(
    const __bf16* __restrict__ qg, const __bf16* __restrict__ kg,
    const __bf16* __restrict__ vT, float* __restrict__ out)
{
  __shared__ __align__(16) __bf16 Klds[32 * 392];   // [key][feat], 25088 B
  __shared__ __align__(16) __bf16 Vtlds[CC * 40];   // [feat][key], 30720 B
  __shared__ __align__(16) __bf16 Plds[8 * 16 * 32];// per-wave P scratch, 8192 B

  const int b     = blockIdx.y;
  const int qbase = blockIdx.x * 64;
  const int tid   = threadIdx.x;
  const int wave  = tid >> 5;
  const int lane  = tid & 31;
  const int l15   = lane & 15;
  const int hi16  = lane >> 4;
  const int p     = wave >> 1;
  const int h     = wave & 1;
  const int qt    = qbase + p * 16;
  const size_t tokb = (size_t)b * TT;
  const float sc  = 0.051031036f * 1.4426950f;      // (1/sqrt(H)) * log2(e)

  // Q fragments for this wave's 16 query rows (full 384-feature K dim).
  v16bf Qa[12];
#pragma unroll
  for (int i = 0; i < 12; ++i) {
    const __bf16* qp = qg + (tokb + qt + l15) * HH + i * 32 + hi16 * 8;
    Qa[i] = cat8(*(const v8bf*)qp, *(const v8bf*)(qp + 16));
  }

  v8f acc[12];
#pragma unroll
  for (int j = 0; j < 12; ++j) acc[j] = (v8f)0.0f;
  float mrow[8], lrow[8];
#pragma unroll
  for (int r = 0; r < 8; ++r) { mrow[r] = -__builtin_inff(); lrow[r] = 0.0f; }

  const int nkt = qbase / 32 + 2;   // causal: key tiles 0 .. qbase+63
  for (int kt = 0; kt < nkt; ++kt) {
    const int kb0 = kt * 32;

    // --- async DMA K tile [32 keys x 384 feats] into LDS (16B per lane) ---
    for (int idx = tid; idx < 32 * 48; idx += 256) {
      int key = idx / 48, off = (idx % 48) * 8;
      async_ld16(&Klds[key * 392 + off],
                 kg + (tokb + kb0 + key) * HH + off);
    }
    // --- async DMA V tile transposed [384 feats x 32 keys] into LDS ---
    for (int idx = tid; idx < CC * 4; idx += 256) {
      int f = idx >> 2, off = (idx & 3) * 8;
      async_ld16(&Vtlds[f * 40 + off],
                 vT + ((size_t)b * HH + f) * TT + kb0 + off);
    }
    // --- prefetch next tile toward L2 while this one is consumed ---
    if (kt + 1 < nkt) {
      const int kn = kb0 + 32;
      __builtin_prefetch(kg + (tokb + kn + (tid >> 3)) * HH + (tid & 7) * 48, 0, 1);
      for (int f = tid; f < CC; f += 256)
        __builtin_prefetch(vT + ((size_t)b * HH + f) * TT + kn, 0, 1);
    }
    wait_async0();      // our async writes to LDS have landed
    __syncthreads();    // everyone's writes visible

    if (kb0 < qt + 16) {            // wave-uniform causal skip (EXEC stays full)
      // S = Q @ Ktile^T : two 16-key C fragments, 24 WMMAs, double-buffered B.
      v8f S0 = (v8f)0.0f, S1 = (v8f)0.0f;
      const __bf16* k0 = &Klds[l15 * 392 + hi16 * 8];
      const __bf16* k1 = &Klds[(16 + l15) * 392 + hi16 * 8];
      v16bf B0 = cat8(*(const v8bf*)k0, *(const v8bf*)(k0 + 16));
      v16bf B1 = cat8(*(const v8bf*)k1, *(const v8bf*)(k1 + 16));
#pragma unroll
      for (int i = 0; i < 12; ++i) {
        v16bf B0n = B0, B1n = B1;
        if (i < 11) {
          const __bf16* n0 = k0 + (i + 1) * 32;
          const __bf16* n1 = k1 + (i + 1) * 32;
          B0n = cat8(*(const v8bf*)n0, *(const v8bf*)(n0 + 16));
          B1n = cat8(*(const v8bf*)n1, *(const v8bf*)(n1 + 16));
        }
        S0 = wmma_bf16(Qa[i], B0, S0);
        S1 = wmma_bf16(Qa[i], B1, S1);
        B0 = B0n; B1 = B1n;
      }

      // mask + scale (exp2 domain) + row max via 16-lane xor reductions
      float mnew[8];
#pragma unroll
      for (int r = 0; r < 8; ++r) {
        int qgl = qt + r + hi16 * 8;
        float s0 = (kb0 + l15      <= qgl) ? S0[r] * sc : -__builtin_inff();
        float s1 = (kb0 + 16 + l15 <= qgl) ? S1[r] * sc : -__builtin_inff();
        S0[r] = s0; S1[r] = s1;
        float mx = fmaxf(s0, s1);
        mx = fmaxf(mx, __shfl_xor(mx, 1));
        mx = fmaxf(mx, __shfl_xor(mx, 2));
        mx = fmaxf(mx, __shfl_xor(mx, 4));
        mx = fmaxf(mx, __shfl_xor(mx, 8));
        mnew[r] = fmaxf(mrow[r], mx);
      }

      // P = exp2(S - m), row sums, accumulator rescale, spill P to LDS
      __bf16* pw = &Plds[wave * 512];
#pragma unroll
      for (int r = 0; r < 8; ++r) {
        float p0 = exp2_fast(S0[r] - mnew[r]);
        float p1 = exp2_fast(S1[r] - mnew[r]);
        int M = r + hi16 * 8;
        pw[M * 32 + l15]      = (__bf16)p0;
        pw[M * 32 + 16 + l15] = (__bf16)p1;
        float rs = p0 + p1;
        rs += __shfl_xor(rs, 1);
        rs += __shfl_xor(rs, 2);
        rs += __shfl_xor(rs, 4);
        rs += __shfl_xor(rs, 8);
        float alpha = exp2_fast(mrow[r] - mnew[r]);
        lrow[r] = lrow[r] * alpha + rs;
        mrow[r] = mnew[r];
#pragma unroll
        for (int j = 0; j < 12; ++j) acc[j][r] *= alpha;
      }

      // Re-read P in A-fragment layout (same-wave DS ordering guarantees RAW)
      const __bf16* pp = &pw[l15 * 32 + hi16 * 8];
      v16bf Pa = cat8(*(const v8bf*)pp, *(const v8bf*)(pp + 16));

      // acc += P(16x32) @ Vtile(32 x 192-half): 12 WMMAs, double-buffered B.
      const __bf16* vb = &Vtlds[(h * 192 + l15) * 40 + hi16 * 8];
      v16bf Bv = cat8(*(const v8bf*)vb, *(const v8bf*)(vb + 16));
#pragma unroll
      for (int j = 0; j < 12; ++j) {
        v16bf Bn = Bv;
        if (j < 11) {
          const __bf16* vp = vb + (size_t)(j + 1) * 16 * 40;
          Bn = cat8(*(const v8bf*)vp, *(const v8bf*)(vp + 16));
        }
        acc[j] = wmma_bf16(Pa, Bv, acc[j]);
        Bv = Bn;
      }
    }
    __syncthreads();   // protect LDS tiles before next stage
  }

  // Epilogue: out = acc / l  (f32 row-major)
#pragma unroll
  for (int r = 0; r < 8; ++r) {
    float inv = 1.0f / lrow[r];
    float* op = out + (tokb + qt + r + hi16 * 8) * HH + h * 192 + l15;
#pragma unroll
    for (int j = 0; j < 12; ++j) op[j * 16] = acc[j][r] * inv;
  }
}

// ---------------------------------------------------------------------------
extern "C" void kernel_launch(void* const* d_in, const int* in_sizes, int n_in,
                              void* d_out, int out_size, void* d_ws, size_t ws_size,
                              hipStream_t stream) {
  const float* x  = (const float*)d_in[0];
  const float* Wq = (const float*)d_in[1];
  const float* Wk = (const float*)d_in[2];
  const float* Wv = (const float*)d_in[3];

  // Workspace: q,k row-major bf16 + v transposed bf16  (3 * 131072*384 * 2 B)
  __bf16* qg = (__bf16*)d_ws;
  __bf16* kg = qg + (size_t)NTOK * HH;
  __bf16* vT = kg + (size_t)NTOK * HH;

  dim3 gp(NTOK / 64, 3);
  qkv_proj_kernel<<<gp, 256, 0, stream>>>(x, Wq, Wk, Wv, qg, kg, vT);

  dim3 ga(TT / 64, NB);
  flash_attn_kernel<<<ga, 256, 0, stream>>>(qg, kg, vT, (float*)d_out);
}
